// rnn_test_40510131536133
// MI455X (gfx1250) — matrix operand order repaired and checked by reading, not compiled, not run on hardware
//
#include <hip/hip_runtime.h>
#include <hip/hip_bf16.h>

#define IN_C    39
#define HID     256
#define BR      8
#define NROW    2048      /* HID*BR */
#define NPADC   296
#define BATCH   32
#define SEQ     200
#define STRIDE_ 10
#define TSTEPS  2000
#define OUTC    39
#define VTH     1.0f

typedef __bf16 bf16;
typedef __attribute__((ext_vector_type(16))) bf16  v16bf;
typedef __attribute__((ext_vector_type(8)))  float v8f;
typedef unsigned int   u32;
typedef unsigned short u16;

__device__ __forceinline__ float sigmoidf_(float x) { return 1.0f / (1.0f + __expf(-x)); }

__device__ __forceinline__ v8f wmma_bf16(v16bf a, v16bf b, v8f c) {
  // v_wmma_f32_16x16x32_bf16: (neg_a, A, neg_b, B, c_mod, C, reuse_a, reuse_b)
  return __builtin_amdgcn_wmma_f32_16x16x32_bf16(false, a, false, b, (short)0, c, false, false);
}

// cross-lane xor-add via DPP quad_perm (stays within quads; masks 1 and 2)
template <int CTRL>
__device__ __forceinline__ float dppAdd(float v) {
  int x = __builtin_bit_cast(int, v);
  int y = __builtin_amdgcn_update_dpp(0, x, CTRL, 0xF, 0xF, true);
  return v + __builtin_bit_cast(float, y);
}

// ---- fragment loaders -------------------------------------------------------
// A (16x32 bf16): lane<16 -> M=lane, K in [0..7]+[16..23]; lane>=16 -> K +8.
__device__ __forceinline__ v16bf loadA_frag(const u16* p /*16B aligned*/) {
  union { uint4 q[2]; v16bf v; } u;
  u.q[0] = *(const uint4*)(p);
  u.q[1] = *(const uint4*)(p + 16);
  return u.v;
}
// B (32x16 bf16), pre-swizzled: frag stored as [lane][8 dwords] -> 2x global b128.
__device__ __forceinline__ v16bf loadB_frag(const u32* p /*32B aligned*/) {
  union { uint4 q[2]; v16bf v; } u;
  u.q[0] = *(const uint4*)(p);
  u.q[1] = *(const uint4*)(p + 4);
  return u.v;
}
// linear B-fragment stream: s2 = chunk*8 + nt*2 + (0=hi,1=lo)
__device__ __forceinline__ const u32* bfragPtr(const u32* bHiD, const u32* bLoD,
                                               int ntBase, int lane, int s2) {
  int chunk = s2 >> 3;
  int ntl   = (s2 >> 1) & 3;
  int hl    = s2 & 1;
  const u32* base = hl ? bLoD : bHiD;
  return base + ((size_t)(((ntBase + ntl) * 8 + chunk) * 32 + lane)) * 8;
}

// ---- prep: masked weights -> bf16 hi/lo, swizzled to WMMA B fragment layout --
__global__ void prep_weights(const float* wf, const float* mf,
                             const float* wb, const float* mb,
                             u16* hiBuf, u16* loBuf) {
  int idx = blockIdx.x * blockDim.x + threadIdx.x;     // 2*2048*256
  if (idx >= 2 * NROW * HID) return;
  int dir = idx >> 19;
  int r   = idx & 524287;
  int n   = r >> 8;                 // output row 0..2047
  int k   = r & 255;                // spike column 0..255
  const float* w = dir ? wb : wf;
  const float* m = dir ? mb : mf;
  float v  = w[n * NPADC + IN_C + k] * m[n * NPADC + IN_C + k];
  bf16 hi  = (bf16)v;
  bf16 lo  = (bf16)(v - (float)hi);
  int ntile = n >> 4, lN = n & 15;
  int chunk = k >> 5, kk = k & 31;
  int lane  = lN + ((kk & 8) ? 16 : 0);
  int kr    = (kk & 7) | ((kk & 16) >> 1);
  int vr    = kr >> 1, half = kr & 1;
  size_t dw = ((((size_t)dir * 128 + ntile) * 8 + chunk) * 32 + lane) * 8 + vr;
  u16 hb, lb;
  __builtin_memcpy(&hb, &hi, 2);
  __builtin_memcpy(&lb, &lo, 2);
  hiBuf[dw * 2 + half] = hb;
  loBuf[dw * 2 + half] = lb;
}

__global__ void prep_scalars(const float* tnf, const float* tnb,
                             const float* tmf, const float* tmb,
                             float* rowBeta, float* alphaH) {
  int t = blockIdx.x * blockDim.x + threadIdx.x;
  if (t < 2 * NROW) {
    const float* s = (t < NROW) ? tnf : tnb;
    rowBeta[t] = sigmoidf_(s[t & (NROW - 1)]);
  }
  if (t < 2 * HID) {
    const float* s = (t < HID) ? tmf : tmb;
    alphaH[t] = sigmoidf_(s[t & (HID - 1)]);
  }
}

// ---- xp[dir][f][b][n] = bias[n] + sum_c x[b,f,c] * (w*mask)[n,c] -------------
__global__ void xproj_kernel(const float* x,
                             const float* wf, const float* mf, const float* bf_,
                             const float* wb, const float* mb, const float* bb_,
                             float* xp) {
  long long idx = (long long)blockIdx.x * blockDim.x + threadIdx.x;
  const long long tot = 2LL * SEQ * BATCH * NROW;
  if (idx >= tot) return;
  int n = (int)(idx & (NROW - 1));
  long long r = idx >> 11;
  int b = (int)(r & 31); r >>= 5;
  int f = (int)(r % SEQ);
  int dir = (int)(r / SEQ);
  const float* w  = dir ? wb : wf;
  const float* m  = dir ? mb : mf;
  const float* bi = dir ? bb_ : bf_;
  const float* xr = x + ((long long)b * SEQ + f) * IN_C;
  const float* wr = w + (long long)n * NPADC;
  const float* mr = m + (long long)n * NPADC;
  float acc = bi[n];
  #pragma unroll
  for (int c = 0; c < IN_C; ++c) acc += xr[c] * wr[c] * mr[c];
  xp[idx] = acc;
}

// ---- per-frame branch sums of xp: sxp[dir][f][b][h] = sum_q xp[...][h*8+q] ---
__global__ void sxp_kernel(const float* xp, float* sxp) {
  long long idx = (long long)blockIdx.x * blockDim.x + threadIdx.x;
  const long long tot = 2LL * SEQ * BATCH * HID;
  if (idx >= tot) return;
  int h = (int)(idx & (HID - 1));
  long long rest = idx >> 8;                 // (dir*SEQ+f)*BATCH + b
  const float* p = xp + rest * NROW + (long long)h * BR;
  float s = 0.f;
  #pragma unroll
  for (int q = 0; q < BR; ++q) s += p[q];
  sxp[idx] = s;
}

// ---- persistent recurrent kernel: 1 block per direction, 32 waves ------------
// Scaled register-resident state: s = (d - xp)/(1-beta) lives in the WMMA
// accumulators. Per step: s' = beta*s + G (VALU prescale, then GEMM chains C
// through s). Branch sums (1-beta)*s reduced cross-lane with DPP quad_perm
// swaps; two quad-partials per neuron land in a small LDS buffer. On frame
// change: s += (xp_old - xp_new)/(1-beta).
__global__ void __launch_bounds__(1024)
rnn_kernel(const u32* bHi, const u32* bLo, const float* xp, const float* sxp,
           const float* rowBeta, const float* alphaH, float* frames) {
  extern __shared__ char smem[];
  u16*   spkA   = (u16*)smem;                          // [32][272] bf16 (17408 B)
  float* alphaS = (float*)(smem + 17408);              // [256]      (1024 B)
  float* sxpS   = (float*)(smem + 18432);              // [32][256]  (32768 B)
  float* dsum2  = (float*)(smem + 51200);              // [32][2][256] (65536 B)
  float* betaS  = (float*)(smem + 116736);             // [2048]     (8192 B)
  float* memS   = (float*)(smem + 124928);             // [32][256]  (32768 B)
  // total 157696 B

  const int dir  = blockIdx.x;
  const int tid  = threadIdx.x;
  const int lane = tid & 31;
  const int wave = tid >> 5;
  const int ntBase = wave * 4;            // 4 N-tiles (64 rows) per wave
  const int lhalf  = lane >> 4;
  const int l15    = lane & 15;

  // init LDS
  for (int i = tid; i < 32 * 272; i += 1024)    spkA[i] = 0;
  for (int i = tid; i < HID; i += 1024)         alphaS[i] = alphaH[dir * HID + i];
  for (int i = tid; i < NROW; i += 1024)        betaS[i] = rowBeta[dir * NROW + i];
  for (int i = tid; i < BATCH * HID; i += 1024) memS[i] = 0.f;

  const u32* bHiD = bHi + (size_t)dir * 128 * 8 * 32 * 8;
  const u32* bLoD = bLo + (size_t)dir * 128 * 8 * 32 * 8;
  const u16* aRow0 = spkA + (size_t)(0  + l15) * 272;   // M-tile 0
  const u16* aRow1 = spkA + (size_t)(16 + l15) * 272;   // M-tile 1

  // phase-C ownership: batch b = tid>>5, neurons h0..h0+7
  const int bIdx = tid >> 5;
  const int h0   = (tid & 31) * 8;
  u32 faccPack = 0u;   // 4-bit j = spike count within current frame (<=10)

  // register-resident scaled dendritic state (the WMMA accumulators)
  v8f sreg[2][4];
  v8f z = {};
  #pragma unroll
  for (int mt = 0; mt < 2; ++mt)
    #pragma unroll
    for (int nt = 0; nt < 4; ++nt) sreg[mt][nt] = z;

  int fprev = -1;
  __syncthreads();

  for (int t = 0; t < TSTEPS; ++t) {
    // input frame: fw -> t/10 ; bw -> ((-t)//10) % 200
    int f = (dir == 0) ? (t / STRIDE_)
                       : ((t == 0) ? 0 : (SEQ - (t + STRIDE_ - 1) / STRIDE_) % SEQ);

    // per-step beta (from LDS; per-lane scalar per N-tile)
    float beV[4];
    #pragma unroll
    for (int nt = 0; nt < 4; ++nt)
      beV[nt] = betaS[(ntBase + nt) * 16 + l15];

    // ---- frame change: rebase s and refresh sxp stage (uniform branch) ----
    if (f != fprev) {
      const float* xpNew = xp + ((size_t)(dir * SEQ + f) * BATCH) * NROW;
      const float* xpOld = (fprev >= 0)
          ? xp + ((size_t)(dir * SEQ + fprev) * BATCH) * NROW : (const float*)0;
      #pragma unroll
      for (int nt = 0; nt < 4; ++nt) {
        float obInv = 1.0f / (1.0f - beV[nt]);
        #pragma unroll
        for (int mt = 0; mt < 2; ++mt) {
          int e0 = (mt * 16 + lhalf * 8) * NROW + (ntBase + nt) * 16 + l15;
          #pragma unroll
          for (int r = 0; r < 8; ++r) {
            int e = e0 + r * NROW;
            float delta = (xpOld ? xpOld[e] : 0.f) - xpNew[e];
            sreg[mt][nt][r] += delta * obInv;
          }
        }
      }
      const float* sxpG = sxp + ((size_t)(dir * SEQ + f) * BATCH) * HID;
      for (int i = tid; i < BATCH * HID; i += 1024) sxpS[i] = sxpG[i];
      fprev = f;
    }

    // ---- decay prescale: s *= beta (state stays in accumulators) ----
    #pragma unroll
    for (int mt = 0; mt < 2; ++mt)
      #pragma unroll
      for (int nt = 0; nt < 4; ++nt)
        #pragma unroll
        for (int r = 0; r < 8; ++r) sreg[mt][nt][r] *= beV[nt];

    // ---- GEMM accumulates G directly into s: s = beta*s + G ----
    v16bf a0, a1;
    v16bf bA = loadB_frag(bfragPtr(bHiD, bLoD, ntBase, lane, 0));
    v16bf bB = loadB_frag(bfragPtr(bHiD, bLoD, ntBase, lane, 1));
    #pragma unroll
    for (int s2 = 0; s2 < 64; ++s2) {
      __builtin_amdgcn_sched_barrier(0x8F);   // pin VMEM to source iteration
      if ((s2 & 7) == 0) {
        int kb = (s2 >> 3) * 32 + lhalf * 8;
        a0 = loadA_frag(aRow0 + kb);
        a1 = loadA_frag(aRow1 + kb);
      }
      v16bf bN = bA;
      if (s2 + 2 < 64) bN = loadB_frag(bfragPtr(bHiD, bLoD, ntBase, lane, s2 + 2));
      int ntl = (s2 >> 1) & 3;
      sreg[0][ntl] = wmma_bf16(a0, bA, sreg[0][ntl]);
      sreg[1][ntl] = wmma_bf16(a1, bA, sreg[1][ntl]);
      bA = bB;
      bB = bN;
    }

    // ---- branch-sum reduction: v = (1-beta)*s, quad_perm xor-adds, two
    //      quad partials per neuron written to dsum2[m][p][h] ----
    #pragma unroll
    for (int mt = 0; mt < 2; ++mt) {
      #pragma unroll
      for (int nt = 0; nt < 4; ++nt) {
        float ob = 1.f - beV[nt];
        int h = (ntBase + nt) * 2 + ((lane >> 3) & 1);
        int p = (lane >> 2) & 1;
        #pragma unroll
        for (int r = 0; r < 8; ++r) {
          float v = ob * sreg[mt][nt][r];
          v = dppAdd<0xB1>(v);                // xor 1 (within quad)
          v = dppAdd<0x4E>(v);                // xor 2 (within quad)
          if ((lane & 3) == 0) {
            int m = mt * 16 + lhalf * 8 + r;
            dsum2[(m * 2 + p) * HID + h] = v;
          }
        }
      }
    }
    __syncthreads();

    // ---- membrane/spike update: thread owns (b, 8 neurons), vector LDS ops ----
    bool flush = ((t % STRIDE_) == (STRIDE_ - 1));
    const float* sxpRow = sxpS + (size_t)bIdx * HID;
    const float* dp0 = dsum2 + ((size_t)bIdx * 2 + 0) * HID;
    const float* dp1 = dsum2 + ((size_t)bIdx * 2 + 1) * HID;
    float* memRow = memS + (size_t)bIdx * HID;
    u16* spkRow = spkA + (size_t)bIdx * 272;

    float4 sxA = *(const float4*)(sxpRow + h0);
    float4 sxB = *(const float4*)(sxpRow + h0 + 4);
    float4 alA = *(const float4*)(alphaS + h0);
    float4 alB = *(const float4*)(alphaS + h0 + 4);
    float4 p0A = *(const float4*)(dp0 + h0);
    float4 p0B = *(const float4*)(dp0 + h0 + 4);
    float4 p1A = *(const float4*)(dp1 + h0);
    float4 p1B = *(const float4*)(dp1 + h0 + 4);
    float4 mA  = *(const float4*)(memRow + h0);
    float4 mB  = *(const float4*)(memRow + h0 + 4);
    float sxv[8] = {sxA.x, sxA.y, sxA.z, sxA.w, sxB.x, sxB.y, sxB.z, sxB.w};
    float alv[8] = {alA.x, alA.y, alA.z, alA.w, alB.x, alB.y, alB.z, alB.w};
    float p0v[8] = {p0A.x, p0A.y, p0A.z, p0A.w, p0B.x, p0B.y, p0B.z, p0B.w};
    float p1v[8] = {p1A.x, p1A.y, p1A.z, p1A.w, p1B.x, p1B.y, p1B.z, p1B.w};
    float mv[8]  = {mA.x, mA.y, mA.z, mA.w, mB.x, mB.y, mB.z, mB.w};

    union { u16 s[8]; uint4 q; } spkOld, spkOut;
    spkOld.q = *(const uint4*)(spkRow + h0);
    #pragma unroll
    for (int j = 0; j < 8; ++j) {
      float s = sxv[j] + p0v[j] + p1v[j];
      float so = spkOld.s[j] ? 1.f : 0.f;
      float mn = mv[j] * alv[j] + (1.f - alv[j]) * s - VTH * so;
      bool  sp = (mn - VTH) > 0.f;
      mv[j] = mn;
      spkOut.s[j] = sp ? (u16)0x3F80 : (u16)0;   // bf16 1.0 / 0.0
      faccPack += (u32)sp << (4 * j);
    }
    *(uint4*)(spkRow + h0) = spkOut.q;
    float4 w0; w0.x = mv[0]; w0.y = mv[1]; w0.z = mv[2]; w0.w = mv[3];
    float4 w1; w1.x = mv[4]; w1.y = mv[5]; w1.z = mv[6]; w1.w = mv[7];
    *(float4*)(memRow + h0)     = w0;
    *(float4*)(memRow + h0 + 4) = w1;

    if (flush) {
      int fr = (dir == 0) ? (t / STRIDE_) : (SEQ - 1 - t / STRIDE_);
      float* fBase = frames + ((size_t)fr * BATCH + bIdx) * (2 * HID) + dir * HID + h0;
      float4 f0, f1;
      f0.x = (float)((faccPack >>  0) & 0xFu) * 0.1f;
      f0.y = (float)((faccPack >>  4) & 0xFu) * 0.1f;
      f0.z = (float)((faccPack >>  8) & 0xFu) * 0.1f;
      f0.w = (float)((faccPack >> 12) & 0xFu) * 0.1f;
      f1.x = (float)((faccPack >> 16) & 0xFu) * 0.1f;
      f1.y = (float)((faccPack >> 20) & 0xFu) * 0.1f;
      f1.z = (float)((faccPack >> 24) & 0xFu) * 0.1f;
      f1.w = (float)((faccPack >> 28) & 0xFu) * 0.1f;
      *(float4*)(fBase)     = f0;
      *(float4*)(fBase + 4) = f1;
      faccPack = 0u;
    }
    __syncthreads();
  }
}

// ---- readout scan + log-softmax + NLL ---------------------------------------
__global__ void __launch_bounds__(256)
readout_kernel(const float* frames, const float* w_ro, const float* b_ro,
               const float* tau, const int* labels, float* out) {
  extern __shared__ char sm[];
  float* wS = (float*)sm;                        // [39][512]
  float* zS = (float*)(sm + OUTC * 512 * 4);     // [32][512]
  __shared__ float yS[BATCH][OUTC];
  __shared__ float memR[BATCH][OUTC];
  __shared__ float lossB[BATCH];
  int tid = threadIdx.x;
  for (int i = tid; i < OUTC * 512; i += 256) wS[i] = w_ro[i];
  for (int i = tid; i < BATCH * OUTC; i += 256) memR[i / OUTC][i % OUTC] = 0.f;
  if (tid < BATCH) lossB[tid] = 0.f;
  __syncthreads();
  for (int fs = 0; fs < SEQ; ++fs) {
    const float* zf = frames + (size_t)fs * BATCH * 2 * HID;
    for (int i = tid; i < BATCH * 512; i += 256) zS[i] = zf[i];
    __syncthreads();
    for (int ij = tid; ij < BATCH * OUTC; ij += 256) {
      int b = ij / OUTC, o = ij - b * OUTC;
      const float* zr = zS + b * 512;
      const float* wr = wS + o * 512;
      float acc = b_ro[o];
      for (int k = 0; k < 512; k += 4)
        acc += zr[k] * wr[k] + zr[k + 1] * wr[k + 1]
             + zr[k + 2] * wr[k + 2] + zr[k + 3] * wr[k + 3];
      float ar = sigmoidf_(tau[o]);
      float mn = memR[b][o] * ar + (1.f - ar) * acc;
      memR[b][o] = mn;
      yS[b][o]   = mn;
    }
    __syncthreads();
    if (tid < BATCH) {
      int b = tid;
      float mx = yS[b][0];
      for (int o = 1; o < OUTC; ++o) mx = fmaxf(mx, yS[b][o]);
      float se = 0.f;
      for (int o = 0; o < OUTC; ++o) se += __expf(yS[b][o] - mx);
      float lse = mx + __logf(se);
      float* op = out + ((size_t)fs * BATCH + b) * OUTC;
      for (int o = 0; o < OUTC; ++o) op[o] = yS[b][o] - lse;
      int lab = labels[b * SEQ + fs];
      lossB[b] -= (yS[b][lab] - lse);
    }
    __syncthreads();
  }
  if (tid == 0) {
    float L = 0.f;
    for (int b = 0; b < BATCH; ++b) L += lossB[b];
    out[(size_t)SEQ * BATCH * OUTC] = L / (float)BATCH;
  }
}

extern "C" void kernel_launch(void* const* d_in, const int* in_sizes, int n_in,
                              void* d_out, int out_size, void* d_ws, size_t ws_size,
                              hipStream_t stream) {
  const float* x       = (const float*)d_in[0];
  const int*   labels  = (const int*)  d_in[1];
  const float* w_fw    = (const float*)d_in[2];
  const float* b_fw    = (const float*)d_in[3];
  const float* tmf     = (const float*)d_in[4];
  const float* tnf     = (const float*)d_in[5];
  const float* mask_fw = (const float*)d_in[6];
  const float* w_bw    = (const float*)d_in[7];
  const float* b_bw    = (const float*)d_in[8];
  const float* tmb     = (const float*)d_in[9];
  const float* tnb     = (const float*)d_in[10];
  const float* mask_bw = (const float*)d_in[11];
  const float* w_ro    = (const float*)d_in[12];
  const float* b_ro    = (const float*)d_in[13];
  const float* tmr     = (const float*)d_in[14];

  char* ws = (char*)d_ws;
  size_t off = 0;
  auto carve = [&](size_t bytes) {
    char* p = ws + off;
    off = (off + bytes + 255) & ~(size_t)255;
    return p;
  };
  u16*   bHi     = (u16*)  carve(2ull * 128 * 8 * 32 * 8 * 4);           // 2 MB
  u16*   bLo     = (u16*)  carve(2ull * 128 * 8 * 32 * 8 * 4);           // 2 MB
  float* rowBeta = (float*)carve((size_t)2 * NROW * 4);
  float* alphaH  = (float*)carve((size_t)2 * HID * 4);
  float* frames  = (float*)carve((size_t)SEQ * BATCH * 2 * HID * 4);     // 13.1 MB
  float* sxp     = (float*)carve(2ull * SEQ * BATCH * HID * 4);          // 13.1 MB
  float* xp      = (float*)carve(2ull * SEQ * BATCH * NROW * 4);         // 104.9 MB

  prep_weights<<<(2 * NROW * HID + 255) / 256, 256, 0, stream>>>(
      w_fw, mask_fw, w_bw, mask_bw, bHi, bLo);
  prep_scalars<<<(2 * NROW + 255) / 256, 256, 0, stream>>>(
      tnf, tnb, tmf, tmb, rowBeta, alphaH);
  long long xtot = 2LL * SEQ * BATCH * NROW;
  xproj_kernel<<<(int)((xtot + 255) / 256), 256, 0, stream>>>(
      x, w_fw, mask_fw, b_fw, w_bw, mask_bw, b_bw, xp);
  long long stot = 2LL * SEQ * BATCH * HID;
  sxp_kernel<<<(int)((stot + 255) / 256), 256, 0, stream>>>(xp, sxp);

  size_t rnnSmem = 157696;   // spkA+alpha+sxp+dsum2+beta+mem
  rnn_kernel<<<2, 1024, rnnSmem, stream>>>(
      (const u32*)bHi, (const u32*)bLo, xp, sxp, rowBeta, alphaH, frames);

  size_t roSmem = (size_t)OUTC * 512 * 4 + (size_t)BATCH * 512 * 4;
  readout_kernel<<<1, 256, roSmem, stream>>>(
      frames, w_ro, b_ro, tmr, labels, (float*)d_out);
}
